// PointTransformerSemSegmentation_56384330662311
// MI455X (gfx1250) — compile-verified, matrix operand order checked by reading, not run on hardware
//
#include <hip/hip_runtime.h>
#include <hip/hip_bf16.h>
#include <math.h>

#define BATCH 4
#define NPTS  4096
#define KNB   16
#define TK    32

typedef __attribute__((ext_vector_type(16))) _Float16 v16h;
typedef __attribute__((ext_vector_type(8)))  float    v8f;
typedef __attribute__((ext_vector_type(4)))  float    f32x4;
typedef __attribute__((ext_vector_type(8)))  _Float16 h8;

static inline int divup(int a, int b) { return (a + b - 1) / b; }

// ============================================================================
// WMMA GEMM: C[M,N] = act(A[M,K] @ W[K,N] + bias [+ C]) ; fp32 in/out,
// f16 WMMA 16x16x32 compute, f32 accumulate.
// Preconditions (hold for every call site in this net): K % 32 == 0,
// M % 64 == 0, A is 16B-aligned with K*4B row stride. N is clamped.
// 256 threads = 8 waves; block tile 64x64; each wave: 16 rows x 32 cols
// (2 accumulators -> 2 v_wmma per K-step, A-fragment reused).
// LDS: A row-major (lane reads 16 contiguous f16 = 2x ds_load_b128),
//      B transposed col-major (same property).
// ============================================================================
__global__ void k_gemm_wmma(const float* __restrict__ A, const float* __restrict__ W,
                            const float* __restrict__ bias, float* __restrict__ C,
                            int M, int N, int K, int act, int addC)
{
    __shared__ _Float16 sA[64 * TK];   // [row][k]
    __shared__ _Float16 sBt[64 * TK];  // [col][k]  (transposed)
    const int tid  = threadIdx.x;
    const int wave = tid >> 5;
    const int lane = tid & 31;
    const int rb = wave & 3;           // row-block 0..3   (16 rows each)
    const int cb = wave >> 2;          // col-block 0..1   (32 cols each)
    const int bm = blockIdx.x * 64, bn = blockIdx.y * 64;
    const int hh = lane >> 4, ml = lane & 15;

    v8f acc0 = {};
    v8f acc1 = {};

    for (int k0 = 0; k0 < K; k0 += TK) {
        // ---- stage A tile 64x32: thread t -> row t/4, 8 cols starting (t%4)*8
        {
            int r = tid >> 2;
            int q = (tid & 3) * 8;
            int gr = bm + r; if (gr >= M) gr = M - 1;      // M%64==0 -> no-op
            const float* src = A + (size_t)gr * K + k0 + q;
            f32x4 v0 = *(const f32x4*)(src);
            f32x4 v1 = *(const f32x4*)(src + 4);
            h8 hv;
            hv[0] = (_Float16)v0[0]; hv[1] = (_Float16)v0[1];
            hv[2] = (_Float16)v0[2]; hv[3] = (_Float16)v0[3];
            hv[4] = (_Float16)v1[0]; hv[5] = (_Float16)v1[1];
            hv[6] = (_Float16)v1[2]; hv[7] = (_Float16)v1[3];
            *(h8*)&sA[r * TK + q] = hv;                    // ds_store_b128
        }
        // ---- stage B tile 32x64 transposed: 512 4-col chunks, 2 per thread
        {
#pragma unroll
            for (int cchunk = 0; cchunk < 2; ++cchunk) {
                int c = tid * 2 + cchunk;
                int k = c >> 4;                 // 0..31
                int nq = (c & 15) * 4;          // 0..60
#pragma unroll
                for (int j = 0; j < 4; ++j) {
                    int gn = bn + nq + j; if (gn >= N) gn = N - 1;
                    float v = W[(size_t)(k0 + k) * N + gn];
                    sBt[(nq + j) * TK + k] = (_Float16)v;
                }
            }
        }
        if (k0 + TK < K) {  // prefetch next K-tile -> global_prefetch_b8
            int pr = bm + lane; if (pr >= M) pr = M - 1;
            __builtin_prefetch((const void*)(A + (size_t)pr * K + k0 + TK), 0, 1);
            __builtin_prefetch((const void*)(W + (size_t)(k0 + TK) * N + bn), 0, 1);
        }
        __syncthreads();
        // ---- fragments: 16 contiguous f16 per lane = 2x ds_load_b128 each
        v16h af  = *(const v16h*)&sA [(rb * 16 + ml) * TK + hh * 16];
        v16h bf0 = *(const v16h*)&sBt[(cb * 32 + ml) * TK + hh * 16];
        v16h bf1 = *(const v16h*)&sBt[(cb * 32 + 16 + ml) * TK + hh * 16];
        acc0 = __builtin_amdgcn_wmma_f32_16x16x32_f16(false, af, false, bf0,
                                                      (short)0, acc0, false, false);
        acc1 = __builtin_amdgcn_wmma_f32_16x16x32_f16(false, af, false, bf1,
                                                      (short)0, acc1, false, false);
        __syncthreads();
    }

    // ---- store: C/D layout: VGPR r -> M = r + (lane>=16)*8, N = lane&15
    const int wm = bm + rb * 16;
#pragma unroll
    for (int half = 0; half < 2; ++half) {
        v8f& acc = half ? acc1 : acc0;
        int n = bn + cb * 32 + half * 16 + ml;
        if (n < N) {
            float bv = bias ? bias[n] : 0.f;
#pragma unroll
            for (int r = 0; r < 8; ++r) {
                int m = wm + r + hh * 8;
                if (m < M) {
                    float v = acc[r] + bv;
                    if (addC) v += C[(size_t)m * N + n];
                    if (act)  v = fmaxf(v, 0.f);
                    C[(size_t)m * N + n] = v;
                }
            }
        }
    }
}

// Scalar GEMM for K not a multiple of 32 (K = 3, 6, 35, 67, 131).
__global__ void k_gemm_small(const float* __restrict__ A, const float* __restrict__ W,
                             const float* __restrict__ bias, float* __restrict__ C,
                             int M, int N, int K, int act, int addC)
{
    int t = blockIdx.x * blockDim.x + threadIdx.x;
    if (t >= M * N) return;
    int m = t / N, n = t % N;
    float s = bias ? bias[n] : 0.f;
    for (int k = 0; k < K; ++k) s += A[(size_t)m * K + k] * W[(size_t)k * N + n];
    if (addC) s += C[t];
    if (act)  s = fmaxf(s, 0.f);
    C[t] = s;
}

// Brute-force 16-NN: one thread per query, insertion-sorted top-16.
__global__ void k_knn16(const float* __restrict__ qx, const float* __restrict__ sx,
                        int* __restrict__ out, int Mq, int Ns)
{
    int b = blockIdx.y;
    int i = blockIdx.x * blockDim.x + threadIdx.x;
    if (i >= Mq) return;
    const float* q = qx + ((size_t)b * Mq + i) * 3;
    const float* S = sx + (size_t)b * Ns * 3;
    float qa = q[0], qb = q[1], qc = q[2];
    float bd[KNB]; int bi[KNB];
    for (int t = 0; t < KNB; ++t) { bd[t] = 3.0e38f; bi[t] = 0; }
    for (int j = 0; j < Ns; ++j) {
        float dx = S[3 * j] - qa, dy = S[3 * j + 1] - qb, dz = S[3 * j + 2] - qc;
        float d = dx * dx + dy * dy + dz * dz;
        if (d < bd[KNB - 1]) {
            int t = KNB - 1;
            while (t > 0 && bd[t - 1] > d) { bd[t] = bd[t - 1]; bi[t] = bi[t - 1]; --t; }
            bd[t] = d; bi[t] = j;
        }
    }
    for (int t = 0; t < KNB; ++t) out[((size_t)b * Mq + i) * KNB + t] = bi[t];
}

// Farthest-point sampling: one 256-thread block per batch.
__global__ void k_fps(const float* __restrict__ xyz, int* __restrict__ sel,
                      float* __restrict__ distws, int Nn, int m)
{
    int b = blockIdx.x;
    const float* X = xyz + (size_t)b * Nn * 3;
    float* dist = distws + (size_t)b * Nn;
    __shared__ float sred[256];
    __shared__ int   sidx[256];
    __shared__ int   sfar;
    for (int i = threadIdx.x; i < Nn; i += blockDim.x) dist[i] = 1e10f;
    if (threadIdx.x == 0) sfar = 0;
    __syncthreads();
    for (int it = 0; it < m; ++it) {
        int f = sfar;
        if (threadIdx.x == 0) sel[(size_t)b * m + it] = f;
        float fx = X[3 * f], fy = X[3 * f + 1], fz = X[3 * f + 2];
        float best = -1.f; int bj = 0;
        for (int i = threadIdx.x; i < Nn; i += blockDim.x) {
            float dx = X[3 * i] - fx, dy = X[3 * i + 1] - fy, dz = X[3 * i + 2] - fz;
            float nd = fminf(dist[i], dx * dx + dy * dy + dz * dz);
            dist[i] = nd;
            if (nd > best) { best = nd; bj = i; }
        }
        sred[threadIdx.x] = best; sidx[threadIdx.x] = bj;
        __syncthreads();
        for (int s = 128; s > 0; s >>= 1) {
            if (threadIdx.x < (unsigned)s && sred[threadIdx.x + s] > sred[threadIdx.x]) {
                sred[threadIdx.x] = sred[threadIdx.x + s];
                sidx[threadIdx.x] = sidx[threadIdx.x + s];
            }
            __syncthreads();
        }
        if (threadIdx.x == 0) sfar = sidx[0];
        __syncthreads();
    }
}

// dst[b,i,j,:] = src[b, idx[b,i,j], :]
__global__ void k_gather(const float* __restrict__ src, const int* __restrict__ idx,
                         float* __restrict__ dst, int Mq, int Kn, int D, int Ns)
{
    int t = blockIdx.x * blockDim.x + threadIdx.x;
    int total = BATCH * Mq * Kn * D;
    if (t >= total) return;
    int c = t % D, r = t / D;
    int j = r % Kn, iq = r / Kn;
    int b = iq / Mq;
    int id = idx[(size_t)iq * Kn + j];
    dst[t] = src[((size_t)b * Ns + id) * D + c];
}

// rel[b,i,j,:] = sgn * (src_xyz[idx] - center_xyz[i])
__global__ void k_rel(const float* __restrict__ center, const float* __restrict__ src,
                      const int* __restrict__ idx, float* __restrict__ out,
                      int Mq, int Ns, float sgn)
{
    int t = blockIdx.x * blockDim.x + threadIdx.x;
    int total = BATCH * Mq * KNB * 3;
    if (t >= total) return;
    int c = t % 3, r = t / 3;
    int j = r % KNB, iq = r / KNB;
    int b = iq / Mq;
    int id = idx[(size_t)iq * KNB + j];
    out[t] = sgn * (src[((size_t)b * Ns + id) * 3 + c] - center[(size_t)iq * 3 + c]);
}

// tmp = q[:,None,:] - kg + pos
__global__ void k_qkp(const float* __restrict__ q, const float* __restrict__ kg,
                      const float* __restrict__ pos, float* __restrict__ tmp,
                      int Mrows, int d)
{
    int t = blockIdx.x * blockDim.x + threadIdx.x;
    int total = Mrows * KNB * d;
    if (t >= total) return;
    int c = t % d;
    int i = (t / d) / KNB;
    tmp[t] = q[(size_t)i * d + c] - kg[t] + pos[t];
}

// softmax over the 16-neighbor axis, per (point,channel)
__global__ void k_softmax_nb(float* __restrict__ a, int Mrows, int d)
{
    int t = blockIdx.x * blockDim.x + threadIdx.x;
    if (t >= Mrows * d) return;
    int i = t / d, c = t % d;
    size_t base = (size_t)i * KNB * d + c;
    float mx = -3.0e38f;
    for (int j = 0; j < KNB; ++j) mx = fmaxf(mx, a[base + (size_t)j * d]);
    float s = 0.f;
    for (int j = 0; j < KNB; ++j) s += expf(a[base + (size_t)j * d] - mx);
    float inv = 1.f / s;
    for (int j = 0; j < KNB; ++j)
        a[base + (size_t)j * d] = expf(a[base + (size_t)j * d] - mx) * inv;
}

// y[i,c] = sum_j a[i,j,c] * (vg[i,j,c] + pos[i,j,c])
__global__ void k_attn_out(const float* __restrict__ a, const float* __restrict__ vg,
                           const float* __restrict__ pos, float* __restrict__ y,
                           int Mrows, int d)
{
    int t = blockIdx.x * blockDim.x + threadIdx.x;
    if (t >= Mrows * d) return;
    int i = t / d, c = t % d;
    size_t base = (size_t)i * KNB * d + c;
    float s = 0.f;
    for (int j = 0; j < KNB; ++j) {
        size_t e = base + (size_t)j * d;
        s += a[e] * (vg[e] + pos[e]);
    }
    y[t] = s;
}

__global__ void k_maxpool_nb(const float* __restrict__ h, float* __restrict__ o,
                             int Mrows, int D)
{
    int t = blockIdx.x * blockDim.x + threadIdx.x;
    if (t >= Mrows * D) return;
    int r = t / D, c = t % D;
    float m = -3.0e38f;
    for (int j = 0; j < KNB; ++j) m = fmaxf(m, h[((size_t)r * KNB + j) * D + c]);
    o[t] = m;
}

__global__ void k_concat2(const float* __restrict__ A, const float* __restrict__ Bb,
                          float* __restrict__ o, int rows, int d1, int d2)
{
    int D = d1 + d2;
    int t = blockIdx.x * blockDim.x + threadIdx.x;
    if (t >= rows * D) return;
    int r = t / D, c = t % D;
    o[t] = (c < d1) ? A[(size_t)r * d1 + c] : Bb[(size_t)r * d2 + (c - d1)];
}

// 3-NN inverse-distance interpolation: out[i,:] += sum_t w_t * f1[idx_t,:]
__global__ void k_interp3(const float* __restrict__ xf, const float* __restrict__ xc,
                          const float* __restrict__ f1, float* __restrict__ out,
                          int nf_, int nc, int co)
{
    int t = blockIdx.x * blockDim.x + threadIdx.x;
    if (t >= BATCH * nf_) return;
    int b = t / nf_;
    const float* q = xf + (size_t)t * 3;
    const float* Xc = xc + (size_t)b * nc * 3;
    float d0 = 3e38f, d1 = 3e38f, d2 = 3e38f;
    int i0 = 0, i1 = 0, i2 = 0;
    for (int j = 0; j < nc; ++j) {
        float dx = Xc[3 * j] - q[0], dy = Xc[3 * j + 1] - q[1], dz = Xc[3 * j + 2] - q[2];
        float d = dx * dx + dy * dy + dz * dz;
        if (d < d0) { d2 = d1; i2 = i1; d1 = d0; i1 = i0; d0 = d; i0 = j; }
        else if (d < d1) { d2 = d1; i2 = i1; d1 = d; i1 = j; }
        else if (d < d2) { d2 = d; i2 = j; }
    }
    float w0 = 1.f / (d0 + 1e-8f), w1 = 1.f / (d1 + 1e-8f), w2 = 1.f / (d2 + 1e-8f);
    float inv = 1.f / (w0 + w1 + w2);
    w0 *= inv; w1 *= inv; w2 *= inv;
    float* orow = out + (size_t)t * co;
    const float* r0 = f1 + ((size_t)b * nc + i0) * co;
    const float* r1 = f1 + ((size_t)b * nc + i1) * co;
    const float* r2 = f1 + ((size_t)b * nc + i2) * co;
    for (int c = 0; c < co; ++c) orow[c] += w0 * r0[c] + w1 * r1[c] + w2 * r2[c];
}

// head: x (b,n,ci) -> out (b,co,n); h = relu(g*(x@c1)+bt); out = h@c2 + b2
__global__ void k_head(const float* __restrict__ x, const float* __restrict__ c1,
                       const float* __restrict__ g, const float* __restrict__ bt,
                       const float* __restrict__ c2, const float* __restrict__ b2,
                       float* __restrict__ out, int n, int ci, int co)
{
    int t = blockIdx.x * blockDim.x + threadIdx.x;
    if (t >= BATCH * n) return;
    int b = t / n, p = t % n;
    const float* xr = x + (size_t)t * ci;
    float h[32];
    for (int d = 0; d < ci; ++d) {
        float s = 0.f;
        for (int c = 0; c < ci; ++c) s += xr[c] * c1[(size_t)c * ci + d];
        s = g[d] * s + bt[d];
        h[d] = s > 0.f ? s : 0.f;
    }
    for (int e = 0; e < co; ++e) {
        float s = b2[e];
        for (int d = 0; d < ci; ++d) s += h[d] * c2[(size_t)d * co + e];
        out[((size_t)b * co + e) * n + p] = s;
    }
}

// L2 normalize over channel axis, x in (b,co,n) layout
__global__ void k_l2norm(float* __restrict__ x, int n, int co)
{
    int t = blockIdx.x * blockDim.x + threadIdx.x;
    if (t >= BATCH * n) return;
    int b = t / n, p = t % n;
    float s = 0.f;
    for (int c = 0; c < co; ++c) {
        float v = x[((size_t)b * co + c) * n + p];
        s += v * v;
    }
    float inv = 1.f / (sqrtf(s) + 1e-12f);
    for (int c = 0; c < co; ++c) x[((size_t)b * co + c) * n + p] *= inv;
}

// bnd[b,p] = sigmoid(max_c edge_preds[b,c,p]) ; ep layout (b,13,n)
__global__ void k_bnd(const float* __restrict__ ep, float* __restrict__ bnd, int n, int nc)
{
    int t = blockIdx.x * blockDim.x + threadIdx.x;
    if (t >= BATCH * n) return;
    int b = t / n, p = t % n;
    float m = -3e38f;
    for (int c = 0; c < nc; ++c) m = fmaxf(m, ep[((size_t)b * nc + c) * n + p]);
    bnd[t] = 1.f / (1.f + expf(-m));
}

__global__ void k_bfm_agg(const float* __restrict__ f, const float* __restrict__ bnd,
                          const float* __restrict__ gm, const int* __restrict__ idxs,
                          float* __restrict__ agg, int n)
{
    int t = blockIdx.x * blockDim.x + threadIdx.x;
    if (t >= BATCH * n) return;
    int b = t / n;
    const int* id = idxs + (size_t)t * KNB;
    const float* gr = gm + (size_t)t * KNB;
    float wv[KNB]; float ssum = 0.f;
    for (int j = 0; j < KNB; ++j) {
        float ww = gr[j] * (1.f - bnd[(size_t)b * n + id[j]]);
        wv[j] = ww; ssum += ww;
    }
    float inv = 1.f / (ssum + 1e-8f);
    for (int c = 0; c < 32; ++c) {
        float acc = 0.f;
        for (int j = 0; j < KNB; ++j)
            acc += wv[j] * f[((size_t)b * n + id[j]) * 32 + c];
        agg[(size_t)t * 32 + c] = acc * inv;
    }
}

__global__ void k_slice3(const float* __restrict__ feat, float* __restrict__ xyz, int total)
{
    int t = blockIdx.x * blockDim.x + threadIdx.x;
    if (t >= total) return;
    xyz[t * 3 + 0] = feat[(size_t)t * 6 + 0];
    xyz[t * 3 + 1] = feat[(size_t)t * 6 + 1];
    xyz[t * 3 + 2] = feat[(size_t)t * 6 + 2];
}

// ============================================================================
// Host-side parameter map (jax pytree: dicts flattened in sorted-key order)
// ============================================================================
struct PtbP { size_t att1, att1_b, att2, att2_b, fc1, fc2, pos1, pos1_b, pos2, pos2_b, wk, wq, wv; };
struct HeadP { size_t b2, bt, c1, c2, g; };
struct TuP { size_t b1, b2, w1, w2; };
struct TdP { size_t b, w; };
struct ParamMap {
    size_t bfm_b, bfm_w;
    HeadP edge_head; PtbP edge_ptb[4]; size_t edge_top; TuP edge_tu[3];
    size_t enc0; PtbP enc_ptb[4];
    HeadP proj, refine_head, seg_head;
    PtbP seg_ptb[4]; size_t seg_top; TuP seg_tu[3];
    TdP td[3];
};
static size_t take(size_t& c, size_t n) { size_t o = c; c += n; return o; }
static PtbP take_ptb(size_t& c, int d) {
    PtbP p;
    p.att1 = take(c, (size_t)d * d); p.att1_b = take(c, d);
    p.att2 = take(c, (size_t)d * d); p.att2_b = take(c, d);
    p.fc1  = take(c, (size_t)d * d); p.fc2    = take(c, (size_t)d * d);
    p.pos1 = take(c, 3 * (size_t)d); p.pos1_b = take(c, d);
    p.pos2 = take(c, (size_t)d * d); p.pos2_b = take(c, d);
    p.wk = take(c, (size_t)d * d); p.wq = take(c, (size_t)d * d); p.wv = take(c, (size_t)d * d);
    return p;
}
static HeadP take_head(size_t& c, int ci, int co) {
    HeadP h; h.b2 = take(c, co); h.bt = take(c, ci); h.c1 = take(c, (size_t)ci * ci);
    h.c2 = take(c, (size_t)ci * co); h.g = take(c, ci); return h;
}
static TuP take_tu(size_t& c, int ci, int co) {
    TuP t; t.b1 = take(c, co); t.b2 = take(c, co);
    t.w1 = take(c, (size_t)ci * co); t.w2 = take(c, (size_t)co * co); return t;
}
static TdP take_td(size_t& c, int din, int dout) {
    TdP t; t.b = take(c, dout); t.w = take(c, (size_t)din * dout); return t;
}
static ParamMap build_params() {
    size_t c = 0; ParamMap m;
    m.bfm_b = take(c, 32); m.bfm_w = take(c, 64 * 32);
    m.edge_head = take_head(c, 32, 13);
    const int dd[4] = {256, 128, 64, 32};
    for (int i = 0; i < 4; ++i) m.edge_ptb[i] = take_ptb(c, dd[i]);
    m.edge_top = take(c, 256 * 256);
    const int tci[3] = {256, 128, 64}, tco[3] = {128, 64, 32};
    for (int i = 0; i < 3; ++i) m.edge_tu[i] = take_tu(c, tci[i], tco[i]);
    m.enc0 = take(c, 6 * 32);
    const int ed[4] = {32, 64, 128, 256};
    for (int i = 0; i < 4; ++i) m.enc_ptb[i] = take_ptb(c, ed[i]);
    m.proj = take_head(c, 32, 32);
    m.refine_head = take_head(c, 32, 13);
    m.seg_head = take_head(c, 32, 13);
    for (int i = 0; i < 4; ++i) m.seg_ptb[i] = take_ptb(c, dd[i]);
    m.seg_top = take(c, 256 * 256);
    for (int i = 0; i < 3; ++i) m.seg_tu[i] = take_tu(c, tci[i], tco[i]);
    const int tdi[3] = {35, 67, 131}, tdo[3] = {64, 128, 256};
    for (int i = 0; i < 3; ++i) m.td[i] = take_td(c, tdi[i], tdo[i]);
    return m;
}

// ---------- workspace bump allocator (deterministic, reset per call) ----------
struct Ws { char* base; size_t off; size_t cap; };
static float* walloc_f(Ws& w, size_t n) {
    float* p = (float*)(w.base + w.off);
    w.off += ((n * sizeof(float)) + 255) & ~(size_t)255;
    return p;
}
static int* walloc_i(Ws& w, size_t n) {
    int* p = (int*)(w.base + w.off);
    w.off += ((n * sizeof(int)) + 255) & ~(size_t)255;
    return p;
}

// ---------- host launch helpers ----------
static void gemm(hipStream_t s, const float* A, const float* W, const float* bias,
                 float* C, int M, int N, int K, int act, int addC) {
    if ((K & 31) == 0) {
        dim3 g(divup(M, 64), divup(N, 64));
        k_gemm_wmma<<<g, 256, 0, s>>>(A, W, bias, C, M, N, K, act, addC);
    } else {
        k_gemm_small<<<divup(M * N, 256), 256, 0, s>>>(A, W, bias, C, M, N, K, act, addC);
    }
}
static void gather(hipStream_t s, const float* src, const int* idx, float* dst,
                   int Mq, int Kn, int D, int Ns) {
    int total = BATCH * Mq * Kn * D;
    k_gather<<<divup(total, 256), 256, 0, s>>>(src, idx, dst, Mq, Kn, D, Ns);
}

// ---------- point-transformer block ----------
static void run_ptb(hipStream_t s, const float* P, const PtbP& pp,
                    const float* feat, const float* xyz, int n, int d,
                    float* out, Ws& w)
{
    size_t mark = w.off;
    int M = BATCH * n;
    int Mn = M * KNB;
    int* idx  = walloc_i(w, (size_t)M * KNB);
    float* x  = walloc_f(w, (size_t)M * d);
    float* q  = walloc_f(w, (size_t)M * d);
    float* kk = walloc_f(w, (size_t)M * d);
    float* v  = walloc_f(w, (size_t)M * d);
    float* y  = walloc_f(w, (size_t)M * d);
    float* rel  = walloc_f(w, (size_t)Mn * 3);
    float* bigA = walloc_f(w, (size_t)Mn * d);
    float* bigB = walloc_f(w, (size_t)Mn * d);
    float* bigC = walloc_f(w, (size_t)Mn * d);
    float* bigD = walloc_f(w, (size_t)Mn * d);

    { dim3 g(divup(n, 64), BATCH); k_knn16<<<g, 64, 0, s>>>(xyz, xyz, idx, n, n); }
    gemm(s, feat, P + pp.fc1, nullptr, x, M, d, d, 0, 0);
    gemm(s, x, P + pp.wq, nullptr, q,  M, d, d, 0, 0);
    gemm(s, x, P + pp.wk, nullptr, kk, M, d, d, 0, 0);
    gemm(s, x, P + pp.wv, nullptr, v,  M, d, d, 0, 0);
    gather(s, kk, idx, bigA, n, KNB, d, n);                                // kg
    gather(s, v,  idx, bigB, n, KNB, d, n);                                // vg
    k_rel<<<divup(BATCH * n * KNB * 3, 256), 256, 0, s>>>(xyz, xyz, idx, rel, n, n, -1.f);
    gemm(s, rel,  P + pp.pos1, P + pp.pos1_b, bigC, Mn, d, 3, 1, 0);       // relu pos-mlp hidden
    gemm(s, bigC, P + pp.pos2, P + pp.pos2_b, bigD, Mn, d, d, 0, 0);       // posf
    k_qkp<<<divup(Mn * d, 256), 256, 0, s>>>(q, bigA, bigD, bigC, M, d);   // tmp
    gemm(s, bigC, P + pp.att1, P + pp.att1_b, bigA, Mn, d, d, 1, 0);       // h = relu(...)
    gemm(s, bigA, P + pp.att2, P + pp.att2_b, bigC, Mn, d, d, 0, 0);       // a
    k_softmax_nb<<<divup(M * d, 256), 256, 0, s>>>(bigC, M, d);
    k_attn_out<<<divup(M * d, 256), 256, 0, s>>>(bigC, bigB, bigD, y, M, d);
    (void)hipMemcpyAsync(out, feat, (size_t)M * d * sizeof(float),
                         hipMemcpyDeviceToDevice, s);
    gemm(s, y, P + pp.fc2, nullptr, out, M, d, d, 0, 1);                   // residual
    w.off = mark;
}

// ---------- transition down ----------
static void run_td(hipStream_t s, const float* P, const TdP& tp,
                   const float* feat, const float* xyz, int n, int din, int m, int dout,
                   float* out_f, float* out_xyz, Ws& w)
{
    size_t mark = w.off;
    int* sel = walloc_i(w, (size_t)BATCH * m);
    float* dist = walloc_f(w, (size_t)BATCH * n);
    int* idx = walloc_i(w, (size_t)BATCH * m * KNB);
    float* nf  = walloc_f(w, (size_t)BATCH * m * KNB * din);
    float* rel = walloc_f(w, (size_t)BATCH * m * KNB * 3);
    float* cat = walloc_f(w, (size_t)BATCH * m * KNB * (din + 3));
    float* h   = walloc_f(w, (size_t)BATCH * m * KNB * dout);
    k_fps<<<BATCH, 256, 0, s>>>(xyz, sel, dist, n, m);
    gather(s, xyz, sel, out_xyz, m, 1, 3, n);
    { dim3 g(divup(m, 64), BATCH); k_knn16<<<g, 64, 0, s>>>(out_xyz, xyz, idx, m, n); }
    gather(s, feat, idx, nf, m, KNB, din, n);
    k_rel<<<divup(BATCH * m * KNB * 3, 256), 256, 0, s>>>(out_xyz, xyz, idx, rel, m, n, 1.f);
    k_concat2<<<divup(BATCH * m * KNB * (din + 3), 256), 256, 0, s>>>(
        nf, rel, cat, BATCH * m * KNB, din, 3);
    gemm(s, cat, P + tp.w, P + tp.b, h, BATCH * m * KNB, dout, din + 3, 1, 0);
    k_maxpool_nb<<<divup(BATCH * m * dout, 256), 256, 0, s>>>(h, out_f, BATCH * m, dout);
    w.off = mark;
}

// ---------- transition up ----------
static void run_tu(hipStream_t s, const float* P, const TuP& tp,
                   const float* f_c, const float* x_c, int nc, int ci,
                   const float* f_f, const float* x_f, int nf_, int co,
                   float* out, Ws& w)
{
    size_t mark = w.off;
    float* f1 = walloc_f(w, (size_t)BATCH * nc * co);
    gemm(s, f_c, P + tp.w1, P + tp.b1, f1, BATCH * nc, co, ci, 0, 0);
    gemm(s, f_f, P + tp.w2, P + tp.b2, out, BATCH * nf_, co, co, 0, 0);
    k_interp3<<<divup(BATCH * nf_, 128), 128, 0, s>>>(x_f, x_c, f1, out, nf_, nc, co);
    w.off = mark;
}

// ---------- full decoder (seg / edge) ----------
static void run_decode(hipStream_t s, const float* P, const PtbP* ptbs, size_t top,
                       const TuP* tus, const float* const fenc[4],
                       const float* const xyzl[4], float* out_f1, Ws& w)
{
    size_t mark = w.off;
    float* t4  = walloc_f(w, (size_t)BATCH * 64 * 256);
    float* df4 = walloc_f(w, (size_t)BATCH * 64 * 256);
    float* t3  = walloc_f(w, (size_t)BATCH * 256 * 128);
    float* df3 = walloc_f(w, (size_t)BATCH * 256 * 128);
    float* t2  = walloc_f(w, (size_t)BATCH * 1024 * 64);
    float* df2 = walloc_f(w, (size_t)BATCH * 1024 * 64);
    float* t1  = walloc_f(w, (size_t)BATCH * 4096 * 32);
    gemm(s, fenc[3], P + top, nullptr, t4, BATCH * 64, 256, 256, 0, 0);
    run_ptb(s, P, ptbs[0], t4, xyzl[3], 64, 256, df4, w);
    run_tu(s, P, tus[0], df4, xyzl[3], 64, 256, fenc[2], xyzl[2], 256, 128, t3, w);
    run_ptb(s, P, ptbs[1], t3, xyzl[2], 256, 128, df3, w);
    run_tu(s, P, tus[1], df3, xyzl[2], 256, 128, fenc[1], xyzl[1], 1024, 64, t2, w);
    run_ptb(s, P, ptbs[2], t2, xyzl[1], 1024, 64, df2, w);
    run_tu(s, P, tus[2], df2, xyzl[1], 1024, 64, fenc[0], xyzl[0], 4096, 32, t1, w);
    run_ptb(s, P, ptbs[3], t1, xyzl[0], 4096, 32, out_f1, w);
    w.off = mark;
}

// ============================================================================
extern "C" void kernel_launch(void* const* d_in, const int* in_sizes, int n_in,
                              void* d_out, int out_size, void* d_ws, size_t ws_size,
                              hipStream_t stream)
{
    (void)in_sizes; (void)n_in; (void)out_size;
    const float* features = (const float*)d_in[0];   // (4,4096,6)
    const float* gmatrix  = (const float*)d_in[1];   // (4,4096,16)
    const int*   idxs     = (const int*)d_in[2];     // (4,4096,16)
    const float* P        = (const float*)d_in[3];   // flattened param pytree
    float* out = (float*)d_out;
    ParamMap pm = build_params();
    Ws w { (char*)d_ws, 0, ws_size };

    // ---- persistent buffers ----
    float* xyz1 = walloc_f(w, (size_t)BATCH * 4096 * 3);
    float* xyz2 = walloc_f(w, (size_t)BATCH * 1024 * 3);
    float* xyz3 = walloc_f(w, (size_t)BATCH * 256 * 3);
    float* xyz4 = walloc_f(w, (size_t)BATCH * 64 * 3);
    float* f1 = walloc_f(w, (size_t)BATCH * 4096 * 32);
    float* f2 = walloc_f(w, (size_t)BATCH * 1024 * 64);
    float* f3 = walloc_f(w, (size_t)BATCH * 256 * 128);
    float* f4 = walloc_f(w, (size_t)BATCH * 64 * 256);
    float* seg_f  = walloc_f(w, (size_t)BATCH * 4096 * 32);
    float* edge_f = walloc_f(w, (size_t)BATCH * 4096 * 32);
    float* refine = walloc_f(w, (size_t)BATCH * 4096 * 32);
    float* catb   = walloc_f(w, (size_t)BATCH * 4096 * 64);
    float* agg    = walloc_f(w, (size_t)BATCH * 4096 * 32);
    float* bnd    = walloc_f(w, (size_t)BATCH * 4096);

    k_slice3<<<divup(BATCH * 4096, 256), 256, 0, stream>>>(features, xyz1, BATCH * 4096);

    // ---- encoder ----
    {
        size_t mk = w.off;
        float* f1p = walloc_f(w, (size_t)BATCH * 4096 * 32);
        gemm(stream, features, P + pm.enc0, nullptr, f1p, BATCH * 4096, 32, 6, 0, 0);
        run_ptb(stream, P, pm.enc_ptb[0], f1p, xyz1, 4096, 32, f1, w);
        w.off = mk;
    }
    {
        size_t mk = w.off;
        float* f2p = walloc_f(w, (size_t)BATCH * 1024 * 64);
        run_td(stream, P, pm.td[0], f1, xyz1, 4096, 32, 1024, 64, f2p, xyz2, w);
        run_ptb(stream, P, pm.enc_ptb[1], f2p, xyz2, 1024, 64, f2, w);
        w.off = mk;
    }
    {
        size_t mk = w.off;
        float* f3p = walloc_f(w, (size_t)BATCH * 256 * 128);
        run_td(stream, P, pm.td[1], f2, xyz2, 1024, 64, 256, 128, f3p, xyz3, w);
        run_ptb(stream, P, pm.enc_ptb[2], f3p, xyz3, 256, 128, f3, w);
        w.off = mk;
    }
    {
        size_t mk = w.off;
        float* f4p = walloc_f(w, (size_t)BATCH * 64 * 256);
        run_td(stream, P, pm.td[2], f3, xyz3, 256, 128, 64, 256, f4p, xyz4, w);
        run_ptb(stream, P, pm.enc_ptb[3], f4p, xyz4, 64, 256, f4, w);
        w.off = mk;
    }

    // ---- decoders ----
    const float* fenc[4] = { f1, f2, f3, f4 };
    const float* xyzl[4] = { xyz1, xyz2, xyz3, xyz4 };
    run_decode(stream, P, pm.seg_ptb,  pm.seg_top,  pm.seg_tu,  fenc, xyzl, seg_f,  w);
    run_decode(stream, P, pm.edge_ptb, pm.edge_top, pm.edge_tu, fenc, xyzl, edge_f, w);

    // ---- heads + BFM + embedding; outputs concatenated flat ----
    const size_t npred = (size_t)BATCH * 13 * 4096;
    float* seg_preds  = out;
    float* seg_ref    = out + npred;
    float* seg_emb    = out + 2 * npred;
    float* edge_preds = out + 2 * npred + (size_t)BATCH * 32 * 4096;

    k_head<<<divup(BATCH * 4096, 128), 128, 0, stream>>>(
        seg_f, P + pm.seg_head.c1, P + pm.seg_head.g, P + pm.seg_head.bt,
        P + pm.seg_head.c2, P + pm.seg_head.b2, seg_preds, 4096, 32, 13);
    k_head<<<divup(BATCH * 4096, 128), 128, 0, stream>>>(
        edge_f, P + pm.edge_head.c1, P + pm.edge_head.g, P + pm.edge_head.bt,
        P + pm.edge_head.c2, P + pm.edge_head.b2, edge_preds, 4096, 32, 13);

    k_bnd<<<divup(BATCH * 4096, 256), 256, 0, stream>>>(edge_preds, bnd, 4096, 13);
    k_bfm_agg<<<divup(BATCH * 4096, 128), 128, 0, stream>>>(seg_f, bnd, gmatrix, idxs, agg, 4096);
    k_concat2<<<divup(BATCH * 4096 * 64, 256), 256, 0, stream>>>(
        seg_f, agg, catb, BATCH * 4096, 32, 32);
    gemm(stream, catb, P + pm.bfm_w, P + pm.bfm_b, refine, BATCH * 4096, 32, 64, 1, 0);
    k_head<<<divup(BATCH * 4096, 128), 128, 0, stream>>>(
        refine, P + pm.refine_head.c1, P + pm.refine_head.g, P + pm.refine_head.bt,
        P + pm.refine_head.c2, P + pm.refine_head.b2, seg_ref, 4096, 32, 13);
    k_head<<<divup(BATCH * 4096, 128), 128, 0, stream>>>(
        seg_f, P + pm.proj.c1, P + pm.proj.g, P + pm.proj.bt,
        P + pm.proj.c2, P + pm.proj.b2, seg_emb, 4096, 32, 32);
    k_l2norm<<<divup(BATCH * 4096, 256), 256, 0, stream>>>(seg_emb, 4096, 32);
}
